// DeformableTransformerEncoderLayer_34127810134137
// MI455X (gfx1250) — matrix operand order repaired and checked by reading, not compiled, not run on hardware
//
#include <hip/hip_runtime.h>

// ---------------- problem constants (match reference) ----------------
#define D_MODEL 256
#define NHEAD   8
#define NLVL    4
#define NPTS    4
#define DFF     1024
#define NB      4
#define LIN     13294                 // 100^2 + 50^2 + 25^2 + 13^2
#define NTOK    (NB * LIN)            // 53176 tokens
#define MPAD    53184                 // NTOK rounded up to multiple of 64

typedef __attribute__((ext_vector_type(16))) _Float16 v16h;
typedef __attribute__((ext_vector_type(8)))  _Float16 v8h;
typedef __attribute__((ext_vector_type(8)))  float    v8f;

union AFrag { v16h v; struct { v8h lo, hi; } s; };

// ---------------- weight convert + transpose: W[K][N] f32 -> Wt[N][K] f16 ----------------
__global__ void wt_convert_kernel(const float* __restrict__ W,
                                  _Float16* __restrict__ Wt, int K, int N) {
    int idx = blockIdx.x * blockDim.x + threadIdx.x;
    if (idx >= K * N) return;
    int k = idx / N, n = idx % N;
    Wt[(size_t)n * K + k] = (_Float16)W[(size_t)k * N + n];
}

// ---------------- q = src + pos ; qh = f16(q) ----------------
__global__ void add_pos_kernel(const float* __restrict__ a, const float* __restrict__ b,
                               float* __restrict__ q, _Float16* __restrict__ qh, int n) {
    int i = blockIdx.x * blockDim.x + threadIdx.x;
    if (i >= n) return;
    float v = a[i] + b[i];
    q[i] = v;
    qh[i] = (_Float16)v;
}

// =====================================================================
// B-resident WMMA GEMM for K == 256.
// Each wave owns a 64-column N-panel: the full B panel (4 tiles x 8 k-steps)
// is loaded into registers ONCE, then the wave loops persistently over
// M-tiles. A fragments are double-buffered so the next k-step's two b128
// loads are in flight behind the current 4-WMMA group.
// =====================================================================
__launch_bounds__(128)
__global__ void gemm_wmma_bres_kernel(const _Float16* __restrict__ A,
                                      const _Float16* __restrict__ Bt,
                                      const float*    __restrict__ bias,
                                      float*          __restrict__ Cf,
                                      _Float16*       __restrict__ Ch,
                                      int M, int N, int relu, int mtStride) {
    const int K    = 256;
    const int lane = threadIdx.x & 31;
    const int wave = threadIdx.x >> 5;
    const int ln   = lane & 15;
    const int hi   = lane >> 4;               // 0 or 1
    const int n0   = blockIdx.y * 64;

    // ---- load the wave's 64xK weight panel into registers (once) ----
    v16h breg[4][8];
#pragma unroll
    for (int t = 0; t < 4; ++t) {
        const _Float16* Bp = Bt + (size_t)(n0 + 16 * t + ln) * K + (hi ? 16 : 0);
#pragma unroll
        for (int ks = 0; ks < 8; ++ks)
            breg[t][ks] = *(const v16h*)(Bp + ks * 32);
    }

    const float bcol[4] = {bias[n0 + 0 + ln],  bias[n0 + 16 + ln],
                           bias[n0 + 32 + ln], bias[n0 + 48 + ln]};

    const int MT = (M + 15) >> 4;
    for (int mt = blockIdx.x * 4 + wave; mt < MT; mt += mtStride) {
        const int m0 = mt * 16;
        int mA = m0 + ln; if (mA > M - 1) mA = M - 1;   // clamp pad rows (never stored)
        const _Float16* Ap = A + (size_t)mA * K + (hi ? 16 : 0);

        v8f acc[4] = {};
        AFrag a0, a1;
        a0.s.lo = *(const v8h*)(Ap + 0);
        a0.s.hi = *(const v8h*)(Ap + 8);
#pragma unroll
        for (int ks = 0; ks < 8; ++ks) {
            if (ks < 7) {                                 // prefetch next fragment
                a1.s.lo = *(const v8h*)(Ap + (ks + 1) * 32);
                a1.s.hi = *(const v8h*)(Ap + (ks + 1) * 32 + 8);
            }
#pragma unroll
            for (int t = 0; t < 4; ++t)
                acc[t] = __builtin_amdgcn_wmma_f32_16x16x32_f16(
                    false, a0.v, false, breg[t][ks], (short)0, acc[t], false, false);
            a0 = a1;
        }

#pragma unroll
        for (int r = 0; r < 8; ++r) {
            int row = m0 + r + (hi ? 8 : 0);
            if (row < M) {
#pragma unroll
                for (int t = 0; t < 4; ++t) {
                    int col = n0 + 16 * t + ln;
                    float v = acc[t][r] + bcol[t];
                    if (relu) v = v > 0.f ? v : 0.f;
                    if (Cf) Cf[(size_t)row * N + col] = v;
                    if (Ch) Ch[(size_t)row * N + col] = (_Float16)v;
                }
            }
        }
    }
}

// =====================================================================
// Streaming WMMA GEMM (any K multiple of 32) — used for ff2 (K=1024),
// whose weight panel is too large to hold in registers.
// =====================================================================
__launch_bounds__(128)
__global__ void gemm_wmma_kernel(const _Float16* __restrict__ A,
                                 const _Float16* __restrict__ Bt,
                                 const float*    __restrict__ bias,
                                 float*          __restrict__ Cf,
                                 _Float16*       __restrict__ Ch,
                                 int M, int N, int K, int relu) {
    const int lane = threadIdx.x & 31;
    const int wave = threadIdx.x >> 5;
    const int ln   = lane & 15;
    const int hi   = lane >> 4;
    const int m0   = blockIdx.x * 64 + wave * 16;
    const int n0   = blockIdx.y * 64;

    int mA = m0 + ln; if (mA > M - 1) mA = M - 1;
    const _Float16* Ap = A + (size_t)mA * K + (hi ? 16 : 0);

    const _Float16* Bp[4];
#pragma unroll
    for (int t = 0; t < 4; ++t)
        Bp[t] = Bt + (size_t)(n0 + 16 * t + ln) * K + (hi ? 16 : 0);

    v8f acc[4] = {};
    for (int k0 = 0; k0 < K; k0 += 32) {
        AFrag a;
        a.s.lo = *(const v8h*)(Ap + k0);
        a.s.hi = *(const v8h*)(Ap + k0 + 8);
#pragma unroll
        for (int t = 0; t < 4; ++t) {
            v16h b = *(const v16h*)(Bp[t] + k0);
            acc[t] = __builtin_amdgcn_wmma_f32_16x16x32_f16(
                false, a.v, false, b, (short)0, acc[t], false, false);
        }
    }

#pragma unroll
    for (int r = 0; r < 8; ++r) {
        int row = m0 + r + (hi ? 8 : 0);
        if (row < M) {
#pragma unroll
            for (int t = 0; t < 4; ++t) {
                int col = n0 + 16 * t + ln;
                float v = acc[t][r] + bias[col];
                if (relu) v = v > 0.f ? v : 0.f;
                if (Cf) Cf[(size_t)row * N + col] = v;
                if (Ch) Ch[(size_t)row * N + col] = (_Float16)v;
            }
        }
    }
}

// ---------------- softmax over the 16 (level,point) logits per (token,head) ----------------
__global__ void softmax16_kernel(float* __restrict__ a, int rows) {
    int r = blockIdx.x * blockDim.x + threadIdx.x;
    if (r >= rows) return;
    float* p = a + (size_t)r * 16;
    float m = p[0];
#pragma unroll
    for (int i = 1; i < 16; ++i) m = fmaxf(m, p[i]);
    float e[16], s = 0.f;
#pragma unroll
    for (int i = 0; i < 16; ++i) { e[i] = __expf(p[i] - m); s += e[i]; }
    float inv = 1.f / s;
#pragma unroll
    for (int i = 0; i < 16; ++i) p[i] = e[i] * inv;
}

// ---------------- deformable attention sampling ----------------
// One wave per (b, query, head); lane = head-dim channel (dh=32).
// Coordinates/weights are uniform across the wave; the 4 bilinear taps per point are
// 32-lane contiguous 128B gathers from value[(b,tok)][head*32 + lane].
__launch_bounds__(256)
__global__ void msda_sample_kernel(const float* __restrict__ value,
                                   const float* __restrict__ offs,
                                   const float* __restrict__ aw,
                                   const float* __restrict__ ref,
                                   _Float16* __restrict__ outh) {
    const int  lane  = threadIdx.x & 31;
    const long gw    = ((long)blockIdx.x * blockDim.x + threadIdx.x) >> 5;
    const long total = (long)NTOK * NHEAD;
    if (gw >= total) return;
    const int  head = (int)(gw % NHEAD);
    const long tok  = gw / NHEAD;              // b*LIN + qi
    const int  b    = (int)(tok / LIN);

    const float* refp = ref  + tok * (NLVL * 2);
    const float* op   = offs + tok * D_MODEL + head * (NLVL * NPTS * 2);
    const float* wp   = aw   + tok * (NHEAD * 16) + head * 16;

    const int Hs[NLVL] = {100, 50, 25, 13};
    const int Ws[NLVL] = {100, 50, 25, 13};
    const int Ss[NLVL] = {0, 10000, 12500, 13125};

    float acc = 0.f;
#pragma unroll
    for (int l = 0; l < NLVL; ++l) {
        const int W = Ws[l], H = Hs[l];
        const float* vbase = value + ((size_t)b * LIN + Ss[l]) * D_MODEL + head * 32 + lane;
        const float rx = refp[l * 2 + 0], ry = refp[l * 2 + 1];
#pragma unroll
        for (int p = 0; p < NPTS; ++p) {
            // loc*W - 0.5 == ref*W + off - 0.5 (normalizer = (W,H))
            float x = rx * (float)W + op[l * (NPTS * 2) + p * 2 + 0] - 0.5f;
            float y = ry * (float)H + op[l * (NPTS * 2) + p * 2 + 1] - 0.5f;
            float x0f = floorf(x), y0f = floorf(y);
            float dx = x - x0f, dy = y - y0f;
            int x0 = (int)x0f, y0 = (int)y0f;
            float w = wp[l * NPTS + p];
            bool xi0 = (x0 >= 0) & (x0 < W);
            bool xi1 = (x0 + 1 >= 0) & (x0 + 1 < W);
            bool yi0 = (y0 >= 0) & (y0 < H);
            bool yi1 = (y0 + 1 >= 0) & (y0 + 1 < H);
            float v00 = 0.f, v10 = 0.f, v01 = 0.f, v11 = 0.f;
            if (yi0 && xi0) v00 = vbase[((size_t)y0 * W + x0) * D_MODEL];
            if (yi0 && xi1) v10 = vbase[((size_t)y0 * W + x0 + 1) * D_MODEL];
            if (yi1 && xi0) v01 = vbase[((size_t)(y0 + 1) * W + x0) * D_MODEL];
            if (yi1 && xi1) v11 = vbase[((size_t)(y0 + 1) * W + x0 + 1) * D_MODEL];
            float s = v00 * (1.f - dx) * (1.f - dy) + v10 * dx * (1.f - dy)
                    + v01 * (1.f - dx) * dy        + v11 * dx * dy;
            acc += w * s;
        }
    }
    outh[tok * D_MODEL + head * 32 + lane] = (_Float16)acc;
}

// ---------------- out = LayerNorm(a + r) * g + beta ; optional f16 copy ----------------
// One wave per row (D=256 -> 8 elems/lane), wave32 butterfly reductions.
__launch_bounds__(256)
__global__ void add_ln_kernel(const float* __restrict__ a, const float* __restrict__ r,
                              const float* __restrict__ g, const float* __restrict__ be,
                              float* __restrict__ outf, _Float16* __restrict__ outh, int M) {
    const int lane = threadIdx.x & 31;
    const int row  = blockIdx.x * 8 + (threadIdx.x >> 5);
    if (row >= M) return;
    const size_t base = (size_t)row * D_MODEL;
    float s[8], sum = 0.f;
#pragma unroll
    for (int t = 0; t < 8; ++t) {
        int j = lane + 32 * t;
        s[t] = a[base + j] + r[base + j];
        sum += s[t];
    }
#pragma unroll
    for (int o = 16; o > 0; o >>= 1) sum += __shfl_xor(sum, o, 32);
    float mean = sum * (1.f / D_MODEL);
    float var = 0.f;
#pragma unroll
    for (int t = 0; t < 8; ++t) { float d = s[t] - mean; var += d * d; }
#pragma unroll
    for (int o = 16; o > 0; o >>= 1) var += __shfl_xor(var, o, 32);
    float rstd = rsqrtf(var * (1.f / D_MODEL) + 1e-5f);
#pragma unroll
    for (int t = 0; t < 8; ++t) {
        int j = lane + 32 * t;
        float v = (s[t] - mean) * rstd * g[j] + be[j];
        outf[base + j] = v;
        if (outh) outh[base + j] = (_Float16)v;
    }
}

// ---------------- host orchestration ----------------
extern "C" void kernel_launch(void* const* d_in, const int* in_sizes, int n_in,
                              void* d_out, int out_size, void* d_ws, size_t ws_size,
                              hipStream_t stream) {
    (void)in_sizes; (void)n_in; (void)out_size; (void)ws_size;
    const float* src     = (const float*)d_in[0];
    const float* pos     = (const float*)d_in[1];
    const float* refpts  = (const float*)d_in[2];
    // d_in[3] spatial_shapes / d_in[4] level_start_index: compile-time constants here
    const float* value_w = (const float*)d_in[5];
    const float* value_b = (const float*)d_in[6];
    const float* offs_w  = (const float*)d_in[7];
    const float* offs_b  = (const float*)d_in[8];
    const float* attn_w  = (const float*)d_in[9];
    const float* attn_b  = (const float*)d_in[10];
    const float* out_w   = (const float*)d_in[11];
    const float* out_b   = (const float*)d_in[12];
    const float* ln1_g   = (const float*)d_in[13];
    const float* ln1_b   = (const float*)d_in[14];
    const float* ff1_w   = (const float*)d_in[15];
    const float* ff1_b   = (const float*)d_in[16];
    const float* ff2_w   = (const float*)d_in[17];
    const float* ff2_b   = (const float*)d_in[18];
    const float* ln2_g   = (const float*)d_in[19];
    const float* ln2_b   = (const float*)d_in[20];

    char* ws = (char*)d_ws;
    size_t off = 0;
    auto alloc = [&](size_t bytes) -> void* {
        void* p = ws + off;
        off += (bytes + 255) & ~(size_t)255;
        return p;
    };

    _Float16* qh    = (_Float16*)alloc((size_t)MPAD * D_MODEL * 2);   // also reused as xh
    _Float16* wt_v  = (_Float16*)alloc((size_t)256 * 256 * 2);
    _Float16* wt_o  = (_Float16*)alloc((size_t)256 * 256 * 2);
    _Float16* wt_a  = (_Float16*)alloc((size_t)128 * 256 * 2);
    _Float16* wt_ow = (_Float16*)alloc((size_t)256 * 256 * 2);
    _Float16* wt_f1 = (_Float16*)alloc((size_t)1024 * 256 * 2);
    _Float16* wt_f2 = (_Float16*)alloc((size_t)256 * 1024 * 2);
    float*    q     = (float*)alloc((size_t)MPAD * D_MODEL * 4);
    float*    value = (float*)alloc((size_t)MPAD * D_MODEL * 4);      // also reused as ff
    float*    offs  = (float*)alloc((size_t)MPAD * D_MODEL * 4);      // also reused as src2
    float*    attn  = (float*)alloc((size_t)MPAD * 128 * 4);          // softmax in-place
    _Float16* msdah = (_Float16*)alloc((size_t)MPAD * D_MODEL * 2);
    float*    x     = (float*)alloc((size_t)MPAD * D_MODEL * 4);
    _Float16* hbuf  = (_Float16*)alloc((size_t)MPAD * DFF * 2);
    float*    ff    = value;   // value is dead after sampling
    float*    src2  = offs;    // offs is dead after sampling
    _Float16* xh    = qh;      // qh is dead after the three q-projections

    // 1) weight convert + transpose
    auto cvt = [&](const float* W, _Float16* Wt, int K, int N) {
        int n = K * N;
        wt_convert_kernel<<<(n + 255) / 256, 256, 0, stream>>>(W, Wt, K, N);
    };
    cvt(value_w, wt_v, 256, 256);
    cvt(offs_w,  wt_o, 256, 256);
    cvt(attn_w,  wt_a, 256, 128);
    cvt(out_w,   wt_ow, 256, 256);
    cvt(ff1_w,   wt_f1, 256, 1024);
    cvt(ff2_w,   wt_f2, 1024, 256);

    // 2) q = src + pos (f32 + f16)
    {
        int n = NTOK * D_MODEL;
        add_pos_kernel<<<(n + 255) / 256, 256, 0, stream>>>(src, pos, q, qh, n);
    }

    // B-resident GEMM (K == 256): persistent M loop, each wave amortizes its
    // register-held 64x256 weight panel over ~6-7 M-tiles.
    const int GX = 128;                       // persistent blocks per N-panel
    auto gemm256 = [&](const _Float16* A, const _Float16* Bt, const float* bias,
                       float* Cf, _Float16* Ch, int N, int relu) {
        dim3 grid(GX, N / 64);
        gemm_wmma_bres_kernel<<<grid, 128, 0, stream>>>(A, Bt, bias, Cf, Ch,
                                                        NTOK, N, relu, GX * 4);
    };

    // 3) projections off q
    gemm256(qh, wt_v, value_b, value, nullptr, 256, 0);   // value
    gemm256(qh, wt_o, offs_b,  offs,  nullptr, 256, 0);   // sampling offsets
    gemm256(qh, wt_a, attn_b,  attn,  nullptr, 128, 0);   // attention logits

    // 4) softmax over 16 per (token, head)
    {
        int rows = NTOK * NHEAD;
        softmax16_kernel<<<(rows + 255) / 256, 256, 0, stream>>>(attn, rows);
    }

    // 5) deformable bilinear sampling -> msda output (f16 for out-projection)
    {
        long waves = (long)NTOK * NHEAD;
        long threads = waves * 32;
        msda_sample_kernel<<<(unsigned)((threads + 255) / 256), 256, 0, stream>>>(
            value, offs, attn, refpts, msdah);
    }

    // 6) output projection: src2 = msda @ out_w + out_b
    gemm256(msdah, wt_ow, out_b, src2, nullptr, 256, 0);

    // 7) x = LN(q + src2)  (f32 + f16 for FFN input)
    {
        int blocks = (NTOK + 7) / 8;
        add_ln_kernel<<<blocks, 256, 0, stream>>>(q, src2, ln1_g, ln1_b, x, xh, NTOK);
    }

    // 8) FFN: h = relu(x @ ff1 + b) in f16 ; ff = h @ ff2 + b in f32
    gemm256(xh, wt_f1, ff1_b, nullptr, hbuf, 1024, 1);
    {
        dim3 grid(MPAD / 64, 256 / 64);       // streaming kernel for K = 1024
        gemm_wmma_kernel<<<grid, 128, 0, stream>>>(hbuf, wt_f2, ff2_b, ff, nullptr,
                                                   NTOK, 256, 1024, 0);
    }

    // 9) out = LN(x + ff)
    {
        int blocks = (NTOK + 7) / 8;
        add_ln_kernel<<<blocks, 256, 0, stream>>>(x, ff, ln2_g, ln2_b, (float*)d_out,
                                                  nullptr, NTOK);
    }
}
